// LightweightESN_AE_81638738362703
// MI455X (gfx1250) — matrix operand order repaired and checked by reading, not compiled, not run on hardware
//
#include <hip/hip_runtime.h>
#include <hip/hip_bf16.h>

// ESN autoencoder, fully fused: 16 waves (one 16-row batch tile each) run the
// whole T=2048 scan with all weights resident as WMMA A-operands.
// State kept transposed (h^T) so it is the WMMA B-operand; C==D layout lets
// biases / chained K-accumulation ride the accumulator.
// Round 3: v_perm_b32-based bf16 pair packing (1 op merge, +0x8000 for
// round-to-nearest) replaces the integer-RNE sequence; hardware v_tanh_f32
// confirmed and kept.

#define Tn 2048
#define Bn 256
#define Dn 64
#define Hn 32
#define Cn 16

typedef __attribute__((ext_vector_type(16))) __bf16          v16bf;
typedef __attribute__((ext_vector_type(2)))  __bf16          v2bf;
typedef __attribute__((ext_vector_type(8)))  float           v8f;
typedef __attribute__((ext_vector_type(8)))  int             v8i;
typedef __attribute__((ext_vector_type(16))) short           v16s;

// ---- bf16 helpers ----
static __device__ __forceinline__ unsigned short f2bf(float f) {
    unsigned u = __float_as_uint(f);
    u += 0x7FFFu + ((u >> 16) & 1u);
    return (unsigned short)(u >> 16);
}

// Pack two f32 into packed bf16 pair {hi,lo}.
static __device__ __forceinline__ int packbf(float lo, float hi) {
#if __has_builtin(__builtin_amdgcn_cvt_pk_bf16_f32)
    v2bf p = __builtin_amdgcn_cvt_pk_bf16_f32(lo, hi);   // v_cvt_pk_bf16_f32
    return __builtin_bit_cast(int, p);
#else
    // round-to-nearest then merge high halves with a single v_perm_b32:
    // D = { uhi[31:24], uhi[23:16], ulo[31:24], ulo[23:16] }  (sel 0x07060302)
    unsigned ulo = __float_as_uint(lo) + 0x8000u;
    unsigned uhi = __float_as_uint(hi) + 0x8000u;
    return (int)__builtin_amdgcn_perm(uhi, ulo, 0x07060302u);
#endif
}

// Hardware tanh (V_TANH_F32 — TRANS op, co-executes with WMMA).
static __device__ __forceinline__ float fast_tanh(float x) {
#if __has_builtin(__builtin_amdgcn_tanhf)
    return __builtin_amdgcn_tanhf(x);                    // v_tanh_f32
#elif __has_builtin(__builtin_amdgcn_tanh_f32)
    return __builtin_amdgcn_tanh_f32(x);
#else
    float e = __builtin_amdgcn_exp2f(x * 2.8853900817779268f);  // exp(2x)
    return 1.0f - 2.0f * __builtin_amdgcn_rcpf(e + 1.0f);
#endif
}

static __device__ __forceinline__ v8f wmma_bf16(v16bf a, v16bf b, v8f c) {
    // v_wmma_f32_16x16x32_bf16  D = A(16x32) * B(32x16) + C
    return __builtin_amdgcn_wmma_f32_16x16x32_bf16(
        /*neg_a=*/false, a, /*neg_b=*/false, b,
        /*c_mod=*/(short)0, c, /*reuse_a=*/false, /*reuse_b=*/false);
}

// A k order used here (16-bit A 16x32, contiguous): lane<16 row M=lane, holds K in {0..15};
// lane>=16 row M=lane-16, K in {16..31}. Element e -> K:
static __device__ __forceinline__ int a_k(int e, int half) {
    return half * 16 + e;
}

// Build an A-operand tile from a row-major fp32 matrix W[m][k], leading dim ldk.
// Prologue-only; runs once per wave.
static __device__ __forceinline__ v16bf load_A(const float* __restrict__ W,
                                               int m, int ldk, int kbase, int half) {
    v16s s;
#pragma unroll
    for (int e = 0; e < 16; ++e)
        s[e] = (short)f2bf(W[m * ldk + kbase + a_k(e, half)]);
    return __builtin_bit_cast(v16bf, s);
}

// B k order used here (16-bit B 32x16, contiguous): lane = column N=lane&15; lanes<16 hold K=0..15,
// lanes>=16 hold K=16..31; dword j = K pair (2j,2j+1) within the half.
// Load 16 consecutive fp32 (this lane's K-run) and pack to bf16.
static __device__ __forceinline__ v16bf load_x16(const float* __restrict__ p) {
    v8i raw;
#pragma unroll
    for (int j = 0; j < 8; ++j) raw[j] = packbf(p[2 * j], p[2 * j + 1]);
    return __builtin_bit_cast(v16bf, raw);
}

// Convert a 32x16 D result (two 16x16 f32 tiles a0=M0..15, a1=M16..31) into the
// next-step B operand: pack pairs to bf16, exchange halves via ds_swizzle SWAPX16.
static __device__ __forceinline__ v16bf to_state(v8f a0, v8f a1, int half) {
    int own0[4], own1[4];
#pragma unroll
    for (int j = 0; j < 4; ++j) {
        own0[j] = packbf(a0[2 * j], a0[2 * j + 1]);   // M = 8*half + 2j,2j+1
        own1[j] = packbf(a1[2 * j], a1[2 * j + 1]);   // M = 16 + 8*half + 2j,2j+1
    }
    v8i raw;
#pragma unroll
    for (int j = 0; j < 4; ++j) {
        int s0 = __builtin_amdgcn_ds_swizzle(own0[j], 0x401F); // SWAPX16
        int s1 = __builtin_amdgcn_ds_swizzle(own1[j], 0x401F);
        raw[j]     = half ? s1 : own0[j];  // K = 16*half + 0..7
        raw[j + 4] = half ? own1[j] : s0;  // K = 16*half + 8..15
    }
    return __builtin_bit_cast(v16bf, raw);
}

__global__ __launch_bounds__(32, 1)
void esn_fused_kernel(const float* __restrict__ x,
                      const float* __restrict__ W_in,   const float* __restrict__ b_in,
                      const float* __restrict__ W_re,   const float* __restrict__ b_re,
                      const float* __restrict__ W_code, const float* __restrict__ b_code,
                      const float* __restrict__ W_co,   const float* __restrict__ b_co,
                      const float* __restrict__ W_rd,   const float* __restrict__ b_rd,
                      const float* __restrict__ W_ro,   const float* __restrict__ b_ro,
                      float* __restrict__ out) {
    const int lane = threadIdx.x & 31;
    const int half = lane >> 4;
    const int col  = lane & 15;       // batch column inside the tile / weight row
    const int b0   = blockIdx.x * 16; // batch tile base

    // ---------- resident A-operand weights ----------
    v16bf A_in[2][2];                 // W_in 32x64: [h-tile][k-chunk]
#pragma unroll
    for (int ht = 0; ht < 2; ++ht)
#pragma unroll
        for (int kc = 0; kc < 2; ++kc)
            A_in[ht][kc] = load_A(W_in, ht * 16 + col, Dn, kc * 32, half);

    v16bf A_re[2], A_rd[2], A_cc[2], A_ro[4];
#pragma unroll
    for (int ht = 0; ht < 2; ++ht) {
        A_re[ht] = load_A(W_re, ht * 16 + col, Hn, 0, half);
        A_rd[ht] = load_A(W_rd, ht * 16 + col, Hn, 0, half);
        // folded code->decode matrix  W_cc = W_co(32x16) @ W_code(16x32)
        v16s s;
#pragma unroll
        for (int e = 0; e < 16; ++e) {
            int m = ht * 16 + col, k = a_k(e, half);
            float acc = 0.f;
#pragma unroll
            for (int c = 0; c < Cn; ++c) acc += W_co[m * Cn + c] * W_code[c * Hn + k];
            s[e] = (short)f2bf(acc);
        }
        A_cc[ht] = __builtin_bit_cast(v16bf, s);
    }
#pragma unroll
    for (int t4 = 0; t4 < 4; ++t4)
        A_ro[t4] = load_A(W_ro, t4 * 16 + col, Hn, 0, half);

    // ---------- biases in C/D layout (element v -> row m = base + 8*half + v) ----------
    v8f CbE[2], CbD[2], CbO[4];
#pragma unroll
    for (int ht = 0; ht < 2; ++ht)
#pragma unroll
        for (int v = 0; v < 8; ++v) {
            int m = ht * 16 + half * 8 + v;
            CbE[ht][v] = b_in[m] + b_re[m];                 // enc: b_in + b_res_enc
            float bcc = b_co[m];                            // dec input: W_co@b_code+b_co
#pragma unroll
            for (int c = 0; c < Cn; ++c) bcc += W_co[m * Cn + c] * b_code[c];
            CbD[ht][v] = bcc + b_rd[m];                     // + b_res_dec
        }
#pragma unroll
    for (int t4 = 0; t4 < 4; ++t4)
#pragma unroll
        for (int v = 0; v < 8; ++v)
            CbO[t4][v] = b_ro[t4 * 16 + half * 8 + v];

    // ---------- zero initial states (B-operand layout) ----------
    v8i z8 = {0, 0, 0, 0, 0, 0, 0, 0};
    v16bf Henc = __builtin_bit_cast(v16bf, z8);
    v16bf Hdec = Henc;

    const float* xrow = x   + (size_t)(b0 + col) * (size_t)Tn * Dn + half * 16;
    float*       orow = out + (size_t)(b0 + col) * (size_t)Tn * Dn + half * 8;

    for (int t = 0; t < Tn; ++t) {
        const float* xp = xrow + (size_t)t * Dn;
        // per-lane contiguous 64B chunks of x -> B operands (K chunks of the input proj)
        v16bf Bx0 = load_x16(xp);        // d = half*16 + e
        v16bf Bx1 = load_x16(xp + 32);   // d = 32 + half*16 + e
        // warm L2/WGP$ for next step's x (off the critical chain)
        __builtin_prefetch(xp + Dn, 0, 3);
        __builtin_prefetch(xp + Dn + 32, 0, 3);

        // u_enc^T + b_in + b_res_enc : 4 chained WMMAs
        v8f a0 = wmma_bf16(A_in[0][0], Bx0, CbE[0]);
        a0     = wmma_bf16(A_in[0][1], Bx1, a0);
        v8f a1 = wmma_bf16(A_in[1][0], Bx0, CbE[1]);
        a1     = wmma_bf16(A_in[1][1], Bx1, a1);
        // + W_res_enc @ h_enc^T
        a0 = wmma_bf16(A_re[0], Henc, a0);
        a1 = wmma_bf16(A_re[1], Henc, a1);
#pragma unroll
        for (int v = 0; v < 8; ++v) { a0[v] = fast_tanh(a0[v]); a1[v] = fast_tanh(a1[v]); }
        Henc = to_state(a0, a1, half);

        // decoder input u_dec^T = W_cc @ enc^T + (b_cc + b_res_dec), then recurrence
        v8f d0 = wmma_bf16(A_cc[0], Henc, CbD[0]);
        v8f d1 = wmma_bf16(A_cc[1], Henc, CbD[1]);
        d0 = wmma_bf16(A_rd[0], Hdec, d0);
        d1 = wmma_bf16(A_rd[1], Hdec, d1);
#pragma unroll
        for (int v = 0; v < 8; ++v) { d0[v] = fast_tanh(d0[v]); d1[v] = fast_tanh(d1[v]); }
        Hdec = to_state(d0, d1, half);

        // readout: out^T(64x16) = W_ro @ h_dec^T + b_ro, stored straight to [B,T,64]
        float* op = orow + (size_t)t * Dn;
#pragma unroll
        for (int t4 = 0; t4 < 4; ++t4) {
            v8f o = wmma_bf16(A_ro[t4], Hdec, CbO[t4]);
            *(float4*)(op + 16 * t4)     = make_float4(o[0], o[1], o[2], o[3]);
            *(float4*)(op + 16 * t4 + 4) = make_float4(o[4], o[5], o[6], o[7]);
        }
    }
}

extern "C" void kernel_launch(void* const* d_in, const int* in_sizes, int n_in,
                              void* d_out, int out_size, void* d_ws, size_t ws_size,
                              hipStream_t stream) {
    (void)in_sizes; (void)n_in; (void)out_size; (void)d_ws; (void)ws_size;
    const float* x      = (const float*)d_in[0];
    const float* W_in   = (const float*)d_in[1];
    const float* b_in   = (const float*)d_in[2];
    const float* W_re   = (const float*)d_in[3];
    const float* b_re   = (const float*)d_in[4];
    const float* W_code = (const float*)d_in[5];
    const float* b_code = (const float*)d_in[6];
    const float* W_co   = (const float*)d_in[7];
    const float* b_co   = (const float*)d_in[8];
    const float* W_rd   = (const float*)d_in[9];
    const float* b_rd   = (const float*)d_in[10];
    const float* W_ro   = (const float*)d_in[11];
    const float* b_ro   = (const float*)d_in[12];
    float* out = (float*)d_out;

    dim3 grid(Bn / 16), block(32);
    hipLaunchKernelGGL(esn_fused_kernel, grid, block, 0, stream,
                       x, W_in, b_in, W_re, b_re, W_code, b_code,
                       W_co, b_co, W_rd, b_rd, W_ro, b_ro, out);
}
